// SAGE_64003602645175
// MI455X (gfx1250) — matrix operand order repaired and checked by reading, not compiled, not run on hardware
//
#include <hip/hip_runtime.h>
#include <hip/hip_bf16.h>

// ---------------------------------------------------------------------------
// GraphSAGE (2-layer) inference for MI455X (gfx1250, wave32, WMMA).
//
// Roofline: ~39 GFLOP of GEMM (tiny at bf16 WMMA rates) vs edge gather
// traffic + 410M fp32 atomics (dominant at 23.3 TB/s). All activations are
// kept bf16-resident (halves gather traffic 1.6 GB -> 0.8 GB, and feeds
// v_wmma_f32_16x16x32_bf16 directly); aggregation accumulates in fp32 via
// L2 atomics. GEMM tiles are staged into double-buffered LDS with the CDNA5
// async global->LDS path (global_load_async_to_lds_b128 / ASYNCcnt), with a
// synchronous fallback that preserves identical addressing.
// ---------------------------------------------------------------------------

typedef __bf16 bf16;
typedef __attribute__((ext_vector_type(4)))  bf16  bf16x4;
typedef __attribute__((ext_vector_type(8)))  bf16  bf16x8;
typedef __attribute__((ext_vector_type(16))) bf16  bf16x16;
typedef __attribute__((ext_vector_type(8)))  float v8f;

// exact parameter type of the async-LDS builtin (from hipcc diagnostic):
// (int4 addrspace(1)*, int4 addrspace(3)*, imm int offset, imm int cpol)
typedef int v4i __attribute__((vector_size(16)));
typedef __attribute__((address_space(1))) v4i* global_v4i;
typedef __attribute__((address_space(3))) v4i* lds_v4i;

#define NODES   50000
#define EDGES   800000
#define IN_DIM  512
#define HID     256
#define OUT_DIM 128

#if __has_builtin(__builtin_amdgcn_global_load_async_to_lds_b128) && \
    __has_builtin(__builtin_amdgcn_s_wait_asynccnt)
#define HAVE_ASYNC_LDS 1
#else
#define HAVE_ASYNC_LDS 0
#endif

// 16-byte global -> LDS copy (async DMA when available).
__device__ __forceinline__ void copy16_to_lds(bf16* l, const bf16* g)
{
#if HAVE_ASYNC_LDS
    __builtin_amdgcn_global_load_async_to_lds_b128(
        (global_v4i)(void*)g, (lds_v4i)(void*)l, /*imm offset=*/0, /*cpol=*/0);
#else
    *(uint4*)l = *(const uint4*)g;
#endif
}

template <int N>
__device__ __forceinline__ void wait_stage()
{
#if HAVE_ASYNC_LDS
    __builtin_amdgcn_s_wait_asynccnt(N);
#endif
}

// ---------------------------------------------------------------------------
// C[M,N] = act( A1 @ W1^T  (+ A2 @ W2^T)  + bias ), all A/W bf16, fp32 acc.
// A row-major [M,K], W row-major [N,K]. Block: 256 threads (8 waves),
// macro-tile 128x64, wave tile 32x32, K-step 32, double-buffered LDS.
// ---------------------------------------------------------------------------
template <bool RELU, bool BF16OUT>
__global__ __launch_bounds__(256)
void wmma_gemm_kernel(const bf16* __restrict__ A1, const bf16* __restrict__ W1,
                      const bf16* __restrict__ A2, const bf16* __restrict__ W2,
                      const float* __restrict__ bias,
                      float* __restrict__ Cf, bf16* __restrict__ Cb,
                      int M, int N, int K)
{
    constexpr int BM = 128, BN = 64, BK = 32;
    constexpr int LDA = 40, LDB = 40;   // padded bf16 strides: 80B, 16B-aligned, conflict-free
    __shared__ bf16 sA[2][BM * LDA];
    __shared__ bf16 sB[2][BN * LDB];

    const int tid   = threadIdx.x;
    const int lane  = tid & 31;
    const int wave  = tid >> 5;
    const int m0    = (wave & 3) * 32;   // 4 waves along M
    const int n0    = (wave >> 2) * 32;  // 2 waves along N
    const int blockM = blockIdx.x * BM;
    const int blockN = blockIdx.y * BN;

    const int half = lane >> 4;          // K-group select per ISA fragment layout
    const int ml   = lane & 15;          // row (A) / col (B,C) within 16

    const int ktiles = K / BK;
    const int nt     = A2 ? 2 * ktiles : ktiles;

    // Stage one 128x32(A) + 64x32(B) bf16 tile: 3 x 16B chunks per thread.
    auto stage = [&](int t, int buf) {
        const bf16* A; const bf16* W; int k0;
        if (t < ktiles) { A = A1; W = W1; k0 = t * BK; }
        else            { A = A2; W = W2; k0 = (t - ktiles) * BK; }
        #pragma unroll
        for (int cc = 0; cc < 2; ++cc) {                 // A tile: 512 chunks
            const int ch = tid + cc * 256;
            const int r = ch >> 2, part = ch & 3;
            int gr = blockM + r; if (gr >= M) gr = M - 1;   // clamp: safe reads
            copy16_to_lds(&sA[buf][r * LDA + part * 8],
                          A + (size_t)gr * K + k0 + part * 8);
        }
        {                                                // B tile: 256 chunks
            const int n = tid >> 2, part = tid & 3;
            copy16_to_lds(&sB[buf][n * LDB + part * 8],
                          W + (size_t)(blockN + n) * K + k0 + part * 8);
        }
    };

    v8f acc[2][2] = {};

    stage(0, 0);
    for (int t = 0; t < nt; ++t) {
        const int cur = t & 1;
        const bool more = (t + 1 < nt);
        if (more) { stage(t + 1, cur ^ 1); wait_stage<3>(); }   // retire tile t, keep t+1 in flight
        else      { wait_stage<0>(); }
        __syncthreads();

        // ---- fragments per documented CDNA5 VGPR layouts ----
        bf16x16 aF[2], bF[2];
        #pragma unroll
        for (int i = 0; i < 2; ++i) {
            // A 16x32: lane half h holds K = h*8+0..7 (VGPR0-3), 16+h*8+0..7 (VGPR4-7)
            const bf16* base = &sA[cur][(m0 + i * 16 + ml) * LDA + half * 8];
            bf16x8 g0 = *(const bf16x8*)(base);
            bf16x8 g1 = *(const bf16x8*)(base + 16);
            aF[i] = __builtin_shufflevector(g0, g1, 0,1,2,3,4,5,6,7,8,9,10,11,12,13,14,15);
        }
        #pragma unroll
        for (int j = 0; j < 2; ++j) {
            // B 32x16: lane half h holds K = h*8+0..7, 16+h*8+0..7 for column n = lane&15
            const bf16* base = &sB[cur][(n0 + j * 16 + ml) * LDB + half * 8];
            bf16x8 g0 = *(const bf16x8*)(base);
            bf16x8 g1 = *(const bf16x8*)(base + 16);
            bF[j] = __builtin_shufflevector(g0, g1, 0,1,2,3,4,5,6,7,8,9,10,11,12,13,14,15);
        }
        #pragma unroll
        for (int i = 0; i < 2; ++i)
            #pragma unroll
            for (int j = 0; j < 2; ++j)
                acc[i][j] = __builtin_amdgcn_wmma_f32_16x16x32_bf16(
                    false, aF[i], false, bF[j], (short)0, acc[i][j], false, false);
        __syncthreads();
    }

    // ---- store: C/D layout VGPR v -> M = v + (lane>=16)*8, N = lane&15 ----
    #pragma unroll
    for (int j = 0; j < 2; ++j) {
        const int col = blockN + n0 + j * 16 + ml;
        const float bv = bias[col];
        #pragma unroll
        for (int i = 0; i < 2; ++i) {
            #pragma unroll
            for (int v = 0; v < 8; ++v) {
                const int row = blockM + m0 + i * 16 + half * 8 + v;
                if (row < M) {
                    float val = acc[i][j][v] + bv;
                    if (RELU) val = fmaxf(val, 0.0f);
                    if (BF16OUT) Cb[(size_t)row * N + col] = (bf16)val;
                    else         Cf[(size_t)row * N + col] = val;
                }
            }
        }
    }
}

// ---------------------------------------------------------------------------
// fp32 -> bf16 (n multiple of 4)
__global__ __launch_bounds__(256)
void f32_to_bf16_kernel(const float* __restrict__ src, bf16* __restrict__ dst, int n)
{
    int i = (blockIdx.x * 256 + threadIdx.x) * 4;
    if (i < n) {
        float4 v = *(const float4*)(src + i);
        bf16x4 o = { (bf16)v.x, (bf16)v.y, (bf16)v.z, (bf16)v.w };
        *(bf16x4*)(dst + i) = o;
    }
}

// mean[i,:] = bf16( agg[i,:] * inv[i] ), cols = HID
__global__ __launch_bounds__(256)
void scale_to_bf16_kernel(const float* __restrict__ agg, const float* __restrict__ inv,
                          bf16* __restrict__ dst, int rows)
{
    int idx = blockIdx.x * 256 + threadIdx.x;
    int r = idx >> 6;                    // HID/4 = 64 groups per row
    int c = (idx & 63) * 4;
    if (r >= rows) return;
    const float s = inv[r];
    float4 v = *(const float4*)&agg[(size_t)r * HID + c];
    bf16x4 o = { (bf16)(v.x * s), (bf16)(v.y * s), (bf16)(v.z * s), (bf16)(v.w * s) };
    *(bf16x4*)&dst[(size_t)r * HID + c] = o;
}

// Degree count: cnt[dst] += 1 (same graph for both conv layers -> run once)
__global__ __launch_bounds__(256)
void scatter_count_kernel(const int* __restrict__ ei, float* __restrict__ cnt, int E)
{
    int e = blockIdx.x * 256 + threadIdx.x;
    if (e < E) atomicAdd(&cnt[ei[E + e]], 1.0f);
}

// cnt -> 1/max(cnt,1) in place
__global__ __launch_bounds__(256)
void invcnt_kernel(float* __restrict__ cnt, int n)
{
    int i = blockIdx.x * 256 + threadIdx.x;
    if (i < n) cnt[i] = 1.0f / fmaxf(cnt[i], 1.0f);
}

// agg[dst,:] += (float)h[src,:] — bf16 gather (8B/thread), fp32 L2 atomics.
__global__ __launch_bounds__(256)
void scatter_feat_kernel(const int* __restrict__ ei, const bf16* __restrict__ h,
                         float* __restrict__ agg, int E)
{
    int idx = blockIdx.x * 256 + threadIdx.x;
    int e   = idx >> 6;                  // 64 groups of 4 per edge (HID=256)
    int fg  = (idx & 63) * 4;
    if (e >= E) return;
    const int s = ei[e];
    const int d = ei[E + e];
    bf16x4 v = *(const bf16x4*)&h[(size_t)s * HID + fg];
    float* dst = &agg[(size_t)d * HID + fg];
    atomicAdd(dst + 0, (float)v[0]);
    atomicAdd(dst + 1, (float)v[1]);
    atomicAdd(dst + 2, (float)v[2]);
    atomicAdd(dst + 3, (float)v[3]);
}

// Row-wise L2 normalize (in place), one wave32 per 128-wide row.
__global__ __launch_bounds__(256)
void l2norm_kernel(float* __restrict__ out, int M, int N)
{
    const int row  = blockIdx.x * 8 + (threadIdx.x >> 5);
    const int lane = threadIdx.x & 31;
    if (row >= M) return;
    float4 v = *(float4*)&out[(size_t)row * N + lane * 4];
    float s = v.x*v.x + v.y*v.y + v.z*v.z + v.w*v.w;
    #pragma unroll
    for (int off = 16; off > 0; off >>= 1) s += __shfl_xor(s, off, 32);
    const float inv = 1.0f / fmaxf(sqrtf(s), 1e-12f);
    v.x *= inv; v.y *= inv; v.z *= inv; v.w *= inv;
    *(float4*)&out[(size_t)row * N + lane * 4] = v;
}

// ---------------------------------------------------------------------------
extern "C" void kernel_launch(void* const* d_in, const int* in_sizes, int n_in,
                              void* d_out, int out_size, void* d_ws, size_t ws_size,
                              hipStream_t stream)
{
    const float* x     = (const float*)d_in[0];
    const int*   ei    = (const int*)  d_in[1];   // [2, E]: row0=src, row1=dst
    const float* Wpre  = (const float*)d_in[2];
    const float* bpre  = (const float*)d_in[3];
    const float* Wpre2 = (const float*)d_in[4];
    const float* bpre2 = (const float*)d_in[5];
    const float* Wl1   = (const float*)d_in[6];
    const float* bl1   = (const float*)d_in[7];
    const float* Wr1   = (const float*)d_in[8];
    const float* Wl2   = (const float*)d_in[9];
    const float* bl2   = (const float*)d_in[10];
    const float* Wr2   = (const float*)d_in[11];
    float* out = (float*)d_out;

    // ---- workspace layout (bytes), with lifetime-based aliasing ----
    char* p = (char*)d_ws;
    const size_t xbBytes   = (size_t)NODES * IN_DIM * 2;     // 51.2 MB
    const size_t aggBytes  = (size_t)NODES * HID * 4;        // 51.2 MB (aliases xb)
    const size_t hbBytes   = (size_t)NODES * HID * 2;        // 25.6 MB
    bf16*  xb    = (bf16*)p;                                  // region0: xb ∪ agg
    float* agg   = (float*)p;
    p += (xbBytes > aggBytes ? xbBytes : aggBytes);
    bf16*  h0b   = (bf16*)p;                                  // region1: h0b ∪ mean
    bf16*  meanb = (bf16*)p;            p += hbBytes;
    bf16*  h1b   = (bf16*)p;            p += hbBytes;
    bf16*  h2b   = (bf16*)p;            p += hbBytes;
    float* cnt   = (float*)p;           p += (size_t)NODES * 4;
    bf16*  wb    = (bf16*)p;                                  // bf16 weights, ~0.8 MB
    bf16*  wb_pre  = wb;
    bf16*  wb_pre2 = wb_pre  + HID * IN_DIM;
    bf16*  wb_l1   = wb_pre2 + HID * HID;
    bf16*  wb_r1   = wb_l1   + HID * HID;
    bf16*  wb_l2   = wb_r1   + HID * HID;
    bf16*  wb_r2   = wb_l2   + OUT_DIM * HID;

    const dim3 blk(256);
    const dim3 gHid((NODES + 127) / 128, HID / 64);
    const dim3 gOut((NODES + 127) / 128, OUT_DIM / 64);
    auto cvtBlocks = [](int n) { return (n / 4 + 255) / 256; };

    // ---- one-time fp32 -> bf16 conversions (weights + input features) ----
    f32_to_bf16_kernel<<<cvtBlocks(HID*IN_DIM),   blk, 0, stream>>>(Wpre,  wb_pre,  HID*IN_DIM);
    f32_to_bf16_kernel<<<cvtBlocks(HID*HID),      blk, 0, stream>>>(Wpre2, wb_pre2, HID*HID);
    f32_to_bf16_kernel<<<cvtBlocks(HID*HID),      blk, 0, stream>>>(Wl1,   wb_l1,   HID*HID);
    f32_to_bf16_kernel<<<cvtBlocks(HID*HID),      blk, 0, stream>>>(Wr1,   wb_r1,   HID*HID);
    f32_to_bf16_kernel<<<cvtBlocks(OUT_DIM*HID),  blk, 0, stream>>>(Wl2,   wb_l2,   OUT_DIM*HID);
    f32_to_bf16_kernel<<<cvtBlocks(OUT_DIM*HID),  blk, 0, stream>>>(Wr2,   wb_r2,   OUT_DIM*HID);
    f32_to_bf16_kernel<<<cvtBlocks(NODES*IN_DIM), blk, 0, stream>>>(x,     xb,      NODES*IN_DIM);

    // ---- feature_pre MLP: h0 = x@Wpre^T+b ; h1 = h0@Wpre2^T+b2 (bf16 out) ----
    wmma_gemm_kernel<false, true><<<gHid, blk, 0, stream>>>(
        xb, wb_pre, nullptr, nullptr, bpre, nullptr, h0b, NODES, HID, IN_DIM);
    wmma_gemm_kernel<false, true><<<gHid, blk, 0, stream>>>(
        h0b, wb_pre2, nullptr, nullptr, bpre2, nullptr, h1b, NODES, HID, HID);

    // ---- degree (shared by both layers) + conv1 aggregation ----
    (void)hipMemsetAsync(agg, 0, aggBytes, stream);
    (void)hipMemsetAsync(cnt, 0, (size_t)NODES * 4, stream);
    scatter_count_kernel<<<(EDGES + 255) / 256, blk, 0, stream>>>(ei, cnt, EDGES);
    scatter_feat_kernel<<<(EDGES * (HID/4) + 255) / 256, blk, 0, stream>>>(ei, h1b, agg, EDGES);
    invcnt_kernel<<<(NODES + 255) / 256, blk, 0, stream>>>(cnt, NODES);
    scale_to_bf16_kernel<<<(NODES * (HID/4) + 255) / 256, blk, 0, stream>>>(agg, cnt, meanb, NODES);

    // ---- conv1: h2 = relu( mean@Wl1^T + bl1 + h1@Wr1^T ) ----
    wmma_gemm_kernel<true, true><<<gHid, blk, 0, stream>>>(
        meanb, wb_l1, h1b, wb_r1, bl1, nullptr, h2b, NODES, HID, HID);

    // ---- conv2 aggregation ----
    (void)hipMemsetAsync(agg, 0, aggBytes, stream);
    scatter_feat_kernel<<<(EDGES * (HID/4) + 255) / 256, blk, 0, stream>>>(ei, h2b, agg, EDGES);
    scale_to_bf16_kernel<<<(NODES * (HID/4) + 255) / 256, blk, 0, stream>>>(agg, cnt, meanb, NODES);

    // ---- conv2: out = mean@Wl2^T + bl2 + h2@Wr2^T (fp32 out) + L2 normalize ----
    wmma_gemm_kernel<false, false><<<gOut, blk, 0, stream>>>(
        meanb, wb_l2, h2b, wb_r2, bl2, out, nullptr, NODES, OUT_DIM, HID);
    l2norm_kernel<<<(NODES + 7) / 8, blk, 0, stream>>>(out, NODES, OUT_DIM);
}
